// MusicAttrFaderNets_14963666059402
// MI455X (gfx1250) — matrix operand order repaired and checked by reading, not compiled, not run on hardware
//
#include <hip/hip_runtime.h>
#include <cstdint>
#include <cstddef>

// ---------------------------------------------------------------------------
// MusicAttrFaderNets on gfx1250 (MI455X), wave32 + v_wmma_f32_16x16x32_f16.
//
// Pipeline (all on `stream`, graph-capturable):
//   1) cvt_pad_kernel x8 + g1_transpose_kernel : f32->f16 weight staging in ws
//   2) enc_kernel (fwd) / enc_kernel (bwd)     : WMMA GRU, LDS-resident h
//   3) latent_kernel                           : mu/std/z/r_out/n_out + Zc/hg1_0
//   4) dec_kernel                              : 2-layer WMMA GRU + log_softmax
//
// Codegen decisions:
//  - Weight fragment addresses are loop-invariant over the 256 time steps;
//    left alone, LICM hoists every fragment and spills KBs/lane to scratch
//    (round 1). Laundering the pointer through asm fixed that but destroyed
//    addrspace provenance -> FLAT loads that tick both LOADcnt and DScnt
//    (round 2). Now we launder only a zero *offset* per step: provenance is
//    kept (global_load_b128 from L2; weights ~12.5 MB << 192 MB L2, shared by
//    all 16 WGs) while the address is still not provably invariant.
//
// Workspace requirement: ~16 MB.
// ---------------------------------------------------------------------------

constexpr int kB    = 256;
constexpr int kT    = 256;
constexpr int kRoll = 342;
constexpr int kHid  = 512;
constexpr int kZd   = 128;
constexpr int kG    = 3 * kHid;   // 1536 gate width
constexpr int kXS   = 352;        // ROLL padded to multiple of 32
constexpr int kHS   = 520;        // LDS h stride (multiple of 8 -> 16B aligned)
constexpr int kG1K  = kRoll + kZd + 2; // 472, g1_w_ih K

typedef __attribute__((ext_vector_type(16))) _Float16 v16h;
typedef __attribute__((ext_vector_type(8)))  _Float16 v8h;
typedef __attribute__((ext_vector_type(8)))  float    v8f;

static __device__ __forceinline__ v8f wmma16(v16h a, v16h b, v8f c) {
  // D = A(16x32 f16) * B(32x16 f16) + C(16x16 f32)
  return __builtin_amdgcn_wmma_f32_16x16x32_f16(
      /*neg_a=*/false, a, /*neg_b=*/false, b,
      /*c_mod=*/(short)0, c, /*reuse_a=*/false, /*reuse_b=*/false);
}

// Opaque zero offset: defeats LICM on weight-fragment loads inside the time
// loop WITHOUT destroying the global address-space provenance of the base
// pointer (so loads lower to global_load_b128, not flat_load_b128).
static __device__ __forceinline__ ptrdiff_t opaque0() {
  ptrdiff_t z = 0;
  asm volatile("" : "+s"(z));
  return z;
}

static __device__ __forceinline__ v8h ld8(const _Float16* p) {
  return *(const v8h*)p;  // 16B aligned by construction -> b128
}

// A fragment: 16x32 f16 tile from row-major LDS [16][strideK].
// Layout here (contiguous, not ISA table order): lane l -> row M = l&15; lanes 0-15 hold K = kc+[0..15],
// lanes 16-31 hold K = kc+[16..31] (contiguous in K, matches frag_b).
static __device__ __forceinline__ v16h frag_a(const _Float16* base, int strideK, int kc) {
  const int l = threadIdx.x & 31;
  const _Float16* p = base + (l & 15) * strideK + kc + ((l >> 4) << 4);
  v8h lo = ld8(p);
  v8h hi = ld8(p + 8);
  return __builtin_shufflevector(lo, hi, 0,1,2,3,4,5,6,7,8,9,10,11,12,13,14,15);
}

// B fragment: 32x16 f16 tile of W^T where W is row-major [N][strideK] in f16.
// Layout here (not ISA table order): lane l -> col N = nt*16 + (l&15); lanes 0-15 hold K = kc+[0..15],
// lanes 16-31 hold K = kc+[16..31] (2 halfs per VGPR, contiguous in K).
static __device__ __forceinline__ v16h frag_b(const _Float16* W, int strideK, int nt, int kc) {
  const int l = threadIdx.x & 31;
  const _Float16* p = W + (size_t)(nt * 16 + (l & 15)) * strideK + kc + ((l >> 4) << 4);
  v8h lo = ld8(p);
  v8h hi = ld8(p + 8);
  return __builtin_shufflevector(lo, hi, 0,1,2,3,4,5,6,7,8,9,10,11,12,13,14,15);
}

static __device__ __forceinline__ float sigm(float x) { return 1.0f / (1.0f + __expf(-x)); }

// ---------------------------------------------------------------------------
// Prep: f32 -> f16 with row/col zero padding. dst is [dstRows][dstStride].
// ---------------------------------------------------------------------------
__global__ void cvt_pad_kernel(_Float16* __restrict__ dst, const float* __restrict__ src,
                               int srcRows, int srcCols, int dstStride, int dstRows) {
  int i = blockIdx.x * blockDim.x + threadIdx.x;
  int total = dstRows * dstStride;
  if (i >= total) return;
  int r = i / dstStride, c = i - r * dstStride;
  float v = (r < srcRows && c < srcCols) ? src[(size_t)r * srcCols + c] : 0.f;
  dst[i] = (_Float16)v;
}

// G1T[c][n] = g1_w_ih[n][c] for c<342 (one-hot feedback columns), f16.
__global__ void g1_transpose_kernel(_Float16* __restrict__ dst, const float* __restrict__ src) {
  int i = blockIdx.x * blockDim.x + threadIdx.x;
  if (i >= kRoll * kG) return;
  int c = i / kG, n = i - c * kG;
  dst[i] = (_Float16)src[(size_t)n * kG1K + c];
}

// ---------------------------------------------------------------------------
// Encoder GRU, one direction. 16 WGs (one per 16-batch-row tile), 512 thr
// = 16 waves. Wave w owns j-tiles {w, w+16} of the 32 hidden 16-col tiles;
// for each it accumulates gate tiles (r: j, u: j+32, n: j+64) for both the
// x-projection (K=352 padded) and the h-projection (K=512) with WMMA.
// Hidden state is double-buffered in LDS (ping-pong on step parity) so the
// gate math writes h_next directly: two barriers per step.
// Only the final hidden state is emitted (scan outputs are unused upstream).
// ---------------------------------------------------------------------------
__global__ __launch_bounds__(512) void enc_kernel(
    const float* __restrict__ x, const _Float16* __restrict__ Wih,
    const _Float16* __restrict__ Whh, const float* __restrict__ bih,
    const float* __restrict__ bhh, float* __restrict__ hout, int backward) {
  __shared__ __align__(16) _Float16 xs[16 * kXS];
  __shared__ __align__(16) _Float16 hs[2][16 * kHS];
  const int b0 = blockIdx.x * 16;
  const int tid = threadIdx.x;
  const int wave = tid >> 5;
  const int l = tid & 31;

  for (int i = tid; i < 16 * kXS; i += 512) xs[i] = (_Float16)0.f;
  for (int i = tid; i < 16 * kHS; i += 512) {
    hs[0][i] = (_Float16)0.f;
    hs[1][i] = (_Float16)0.f;
  }
  __syncthreads();

  for (int step = 0; step < kT; ++step) {
    const int t = backward ? (kT - 1 - step) : step;
    const int cur = step & 1;
    const _Float16* hc = hs[cur];
    _Float16* hn = hs[cur ^ 1];
    const ptrdiff_t zz = opaque0();   // per-step: block LICM of weight frags
    const _Float16* wih = Wih + zz;
    const _Float16* whh = Whh + zz;

    for (int i = tid; i < 16 * kRoll; i += 512) {
      int m = i / kRoll, c = i - m * kRoll;
      xs[m * kXS + c] = (_Float16)x[((size_t)(b0 + m) * kT + t) * kRoll + c];
    }
    __syncthreads();

#pragma unroll
    for (int q = 0; q < 2; ++q) {
      const int jt = wave + 16 * q;  // 0..31
      v8f giR = {}, giU = {}, giN = {}, ghR = {}, ghU = {}, ghN = {};
      for (int kc = 0; kc < kXS; kc += 32) {
        v16h a = frag_a(xs, kXS, kc);
        giR = wmma16(a, frag_b(wih, kXS, jt, kc), giR);
        giU = wmma16(a, frag_b(wih, kXS, jt + 32, kc), giU);
        giN = wmma16(a, frag_b(wih, kXS, jt + 64, kc), giN);
      }
      for (int kc = 0; kc < kHid; kc += 32) {
        v16h a = frag_a(hc, kHS, kc);
        ghR = wmma16(a, frag_b(whh, kHid, jt, kc), ghR);
        ghU = wmma16(a, frag_b(whh, kHid, jt + 32, kc), ghU);
        ghN = wmma16(a, frag_b(whh, kHid, jt + 64, kc), ghN);
      }
      const int jc = jt * 16 + (l & 15);
      const float bir = bih[jc], biu = bih[kHid + jc], bin = bih[2 * kHid + jc];
      const float bhr = bhh[jc], bhu = bhh[kHid + jc], bhn = bhh[2 * kHid + jc];
#pragma unroll
      for (int g = 0; g < 8; ++g) {
        const int m = g + ((l >> 4) << 3);  // C/D layout: M = vgpr + 8*(lane/16)
        float r  = sigm(giR[g] + bir + ghR[g] + bhr);
        float u  = sigm(giU[g] + biu + ghU[g] + bhu);
        float nn = tanhf(giN[g] + bin + r * (ghN[g] + bhn));
        float ho = (float)hc[m * kHS + jc];
        hn[m * kHS + jc] = (_Float16)((1.f - u) * nn + u * ho);
      }
    }
    __syncthreads();  // hn complete + xs reads done before next overwrite
  }

  const _Float16* hfin = hs[kT & 1];  // kT even -> buffer 0
  for (int i = tid; i < 16 * kHid; i += 512) {
    int m = i / kHid, j = i - m * kHid;
    hout[(size_t)(b0 + m) * kHid + j] = (float)hfin[m * kHS + j];
  }
}

// ---------------------------------------------------------------------------
// Latent head: mu, std, z0, z, r_out, n_out; precomputes decoder constants
// Zc[b][1536] = z @ g1_w_ih[:,342:]^T + g1_b_ih and hg1_0 = z @ w_init^T + b.
// Small (~135 MFLOP total) -> scalar VALU kernel, one block per batch row.
// ---------------------------------------------------------------------------
__global__ __launch_bounds__(256) void latent_kernel(
    const float* __restrict__ hf, const float* __restrict__ hb,
    const float* __restrict__ w_mu, const float* __restrict__ b_mu,
    const float* __restrict__ w_var, const float* __restrict__ b_var,
    const float* __restrict__ w_dr, const float* __restrict__ b_dr,
    const float* __restrict__ w_dn, const float* __restrict__ b_dn,
    const float* __restrict__ w_init, const float* __restrict__ b_init,
    const float* __restrict__ g1_w_ih, const float* __restrict__ g1_b_ih,
    const float* __restrict__ eps, const float* __restrict__ rd,
    const float* __restrict__ nd,
    float* __restrict__ o_mu, float* __restrict__ o_std, float* __restrict__ o_z,
    float* __restrict__ o_r, float* __restrict__ o_n,
    float* __restrict__ Zc, float* __restrict__ hg10) {
  __shared__ float hsh[2 * kHid];
  __shared__ float zsh[kZd + 2];
  const int b = blockIdx.x, tid = threadIdx.x;
  for (int i = tid; i < kHid; i += 256) {
    hsh[i]        = hf[(size_t)b * kHid + i];
    hsh[kHid + i] = hb[(size_t)b * kHid + i];
  }
  __syncthreads();
  if (tid < kZd) {
    const float* wm = w_mu + (size_t)tid * (2 * kHid);
    const float* wv = w_var + (size_t)tid * (2 * kHid);
    float sm = 0.f, sv = 0.f;
    for (int k = 0; k < 2 * kHid; ++k) { sm += hsh[k] * wm[k]; sv += hsh[k] * wv[k]; }
    float mu = sm + b_mu[tid];
    float st = __expf(sv + b_var[tid]);
    float z0 = mu + st * eps[(size_t)b * kZd + tid];
    o_mu[(size_t)b * kZd + tid] = mu;
    o_std[(size_t)b * kZd + tid] = st;
    zsh[tid] = z0;
    o_z[(size_t)b * (kZd + 2) + tid] = z0;
  } else if (tid == kZd) {
    zsh[kZd] = rd[b];
    o_z[(size_t)b * (kZd + 2) + kZd] = rd[b];
  } else if (tid == kZd + 1) {
    zsh[kZd + 1] = nd[b];
    o_z[(size_t)b * (kZd + 2) + kZd + 1] = nd[b];
  }
  __syncthreads();
  if (tid == 0) {
    float s = b_dr[0];
    for (int k = 0; k < kZd; ++k) s += zsh[k] * w_dr[k];
    o_r[b] = fmaxf(s, 0.f);
  } else if (tid == 1) {
    float s = b_dn[0];
    for (int k = 0; k < kZd; ++k) s += zsh[k] * w_dn[k];
    o_n[b] = fmaxf(s, 0.f);
  }
  for (int j = tid; j < kHid; j += 256) {
    const float* wi = w_init + (size_t)j * (kZd + 2);
    float s = b_init[j];
    for (int k = 0; k < kZd + 2; ++k) s += zsh[k] * wi[k];
    hg10[(size_t)b * kHid + j] = s;
  }
  for (int n = tid; n < kG; n += 256) {
    const float* wr = g1_w_ih + (size_t)n * kG1K + kRoll;
    float s = g1_b_ih[n];
    for (int k = 0; k < kZd + 2; ++k) s += zsh[k] * wr[k];
    Zc[(size_t)b * kG + n] = s;
  }
}

// ---------------------------------------------------------------------------
// Autoregressive 2-layer GRU decoder. 16 WGs x 512 threads.
// One-hot feedback -> zero-FLOP row gather from G1T + precomputed Zc.
// Three K=512 WMMA GEMMs + one K=512 output WMMA GEMM per step.
// Hidden states double-buffered in LDS (ping-pong on t parity).
// ---------------------------------------------------------------------------
__global__ __launch_bounds__(512) void dec_kernel(
    const _Float16* __restrict__ G1hh, const _Float16* __restrict__ G2ih,
    const _Float16* __restrict__ G2hh, const _Float16* __restrict__ G1T,
    const _Float16* __restrict__ Wout,
    const float* __restrict__ Zc, const float* __restrict__ hg10,
    const float* __restrict__ g1_bhh,
    const float* __restrict__ g2_bih, const float* __restrict__ g2_bhh,
    const float* __restrict__ b_out, float* __restrict__ out) {
  __shared__ __align__(16) _Float16 h1s[2][16 * kHS];
  __shared__ __align__(16) _Float16 h2s[2][16 * kHS];
  __shared__ float lg[16 * kXS];
  __shared__ int   cidx[16];
  __shared__ float rowoff[16];
  const int b0 = blockIdx.x * 16;
  const int tid = threadIdx.x;
  const int wave = tid >> 5;
  const int l = tid & 31;

  for (int i = tid; i < 16 * kHS; i += 512) {
    h1s[0][i] = (_Float16)0.f; h1s[1][i] = (_Float16)0.f;
    h2s[0][i] = (_Float16)0.f; h2s[1][i] = (_Float16)0.f;
  }
  __syncthreads();
  for (int i = tid; i < 16 * kHid; i += 512) {
    int m = i / kHid, j = i - m * kHid;
    h1s[0][m * kHS + j] = (_Float16)hg10[(size_t)(b0 + m) * kHid + j];
  }
  if (tid < 16) cidx[tid] = kRoll - 1;  // out0 one-hot at last index
  __syncthreads();

  for (int t = 0; t < kT; ++t) {
    const int cur = t & 1;
    const _Float16* h1c = h1s[cur];
    _Float16* h1n = h1s[cur ^ 1];
    _Float16* h2c = h2s[cur];
    _Float16* h2n = h2s[cur ^ 1];
    const ptrdiff_t zz = opaque0();   // per-step: block LICM of weight frags
    const _Float16* g1hh = G1hh + zz;
    const _Float16* g2ih = G2ih + zz;
    const _Float16* g2hh = G2hh + zz;
    const _Float16* wout = Wout + zz;
    const float*    zc   = Zc + zz;

    // ---- GRU layer 1 ----
#pragma unroll
    for (int q = 0; q < 2; ++q) {
      const int jt = wave + 16 * q;
      v8f ghR = {}, ghU = {}, ghN = {};
      for (int kc = 0; kc < kHid; kc += 32) {
        v16h a = frag_a(h1c, kHS, kc);
        ghR = wmma16(a, frag_b(g1hh, kHid, jt, kc), ghR);
        ghU = wmma16(a, frag_b(g1hh, kHid, jt + 32, kc), ghU);
        ghN = wmma16(a, frag_b(g1hh, kHid, jt + 64, kc), ghN);
      }
      const int jc = jt * 16 + (l & 15);
      const float bhr = g1_bhh[jc], bhu = g1_bhh[kHid + jc], bhn = g1_bhh[2 * kHid + jc];
#pragma unroll
      for (int g = 0; g < 8; ++g) {
        const int m = g + ((l >> 4) << 3);
        const int c = cidx[m];
        const size_t zb = (size_t)(b0 + m) * kG;
        const size_t wb = (size_t)c * kG;
        float giR = (float)G1T[wb + jc]            + zc[zb + jc];
        float giU = (float)G1T[wb + kHid + jc]     + zc[zb + kHid + jc];
        float giN = (float)G1T[wb + 2 * kHid + jc] + zc[zb + 2 * kHid + jc];
        float r  = sigm(giR + ghR[g] + bhr);
        float u  = sigm(giU + ghU[g] + bhu);
        float nn = tanhf(giN + r * (ghN[g] + bhn));
        float ho = (float)h1c[m * kHS + jc];
        h1n[m * kHS + jc] = (_Float16)((1.f - u) * nn + u * ho);
      }
    }
    __syncthreads();
    if (t == 0) {  // hx[1] seeded with hx[0] at step 0
      for (int i = tid; i < 16 * kHS; i += 512) h2c[i] = h1n[i];
      __syncthreads();
    }
    // ---- GRU layer 2 ----
#pragma unroll
    for (int q = 0; q < 2; ++q) {
      const int jt = wave + 16 * q;
      v8f giR = {}, giU = {}, giN = {}, ghR = {}, ghU = {}, ghN = {};
      for (int kc = 0; kc < kHid; kc += 32) {
        v16h a1 = frag_a(h1n, kHS, kc);
        giR = wmma16(a1, frag_b(g2ih, kHid, jt, kc), giR);
        giU = wmma16(a1, frag_b(g2ih, kHid, jt + 32, kc), giU);
        giN = wmma16(a1, frag_b(g2ih, kHid, jt + 64, kc), giN);
        v16h a2 = frag_a(h2c, kHS, kc);
        ghR = wmma16(a2, frag_b(g2hh, kHid, jt, kc), ghR);
        ghU = wmma16(a2, frag_b(g2hh, kHid, jt + 32, kc), ghU);
        ghN = wmma16(a2, frag_b(g2hh, kHid, jt + 64, kc), ghN);
      }
      const int jc = jt * 16 + (l & 15);
      const float bir = g2_bih[jc], biu = g2_bih[kHid + jc], bin = g2_bih[2 * kHid + jc];
      const float bhr = g2_bhh[jc], bhu = g2_bhh[kHid + jc], bhn = g2_bhh[2 * kHid + jc];
#pragma unroll
      for (int g = 0; g < 8; ++g) {
        const int m = g + ((l >> 4) << 3);
        float r  = sigm(giR[g] + bir + ghR[g] + bhr);
        float u  = sigm(giU[g] + biu + ghU[g] + bhu);
        float nn = tanhf(giN[g] + bin + r * (ghN[g] + bhn));
        float ho = (float)h2c[m * kHS + jc];
        h2n[m * kHS + jc] = (_Float16)((1.f - u) * nn + u * ho);
      }
    }
    __syncthreads();
    // ---- output projection (22 n-tiles over padded 352) ----
    for (int nt = wave; nt < 22; nt += 16) {
      v8f acc = {};
      for (int kc = 0; kc < kHid; kc += 32)
        acc = wmma16(frag_a(h2n, kHS, kc), frag_b(wout, kHid, nt, kc), acc);
      const int n = nt * 16 + (l & 15);
      const float bo = (n < kRoll) ? b_out[n] : 0.f;
#pragma unroll
      for (int g = 0; g < 8; ++g)
        lg[(g + ((l >> 4) << 3)) * kXS + n] = acc[g] + bo;
    }
    __syncthreads();
    // ---- log_softmax stats + argmax feedback ----
    if (tid < 16) {
      const int m = tid;
      float mx = -3.4e38f; int am = 0;
      for (int c = 0; c < kRoll; ++c) {
        float v = lg[m * kXS + c];
        if (v > mx) { mx = v; am = c; }
      }
      float s = 0.f;
      for (int c = 0; c < kRoll; ++c) s += __expf(lg[m * kXS + c] - mx);
      rowoff[m] = mx + __logf(s);
      cidx[m] = am;
    }
    __syncthreads();
    for (int i = tid; i < 16 * kRoll; i += 512) {
      int m = i / kRoll, c = i - m * kRoll;
      out[((size_t)(b0 + m) * kT + t) * kRoll + c] = lg[m * kXS + c] - rowoff[m];
    }
    __syncthreads();
  }
}

// ---------------------------------------------------------------------------
extern "C" void kernel_launch(void* const* d_in, const int* in_sizes, int n_in,
                              void* d_out, int out_size, void* d_ws, size_t ws_size,
                              hipStream_t stream) {
  (void)in_sizes; (void)n_in; (void)out_size; (void)ws_size;
  // setup_inputs order: x, rhythm, note, chroma, r_density, n_density, eps,
  // then params dict in insertion order.
  const float* x         = (const float*)d_in[0];
  const float* r_density = (const float*)d_in[4];
  const float* n_density = (const float*)d_in[5];
  const float* eps       = (const float*)d_in[6];
  const float* eWihF = (const float*)d_in[7];
  const float* eWhhF = (const float*)d_in[8];
  const float* eBihF = (const float*)d_in[9];
  const float* eBhhF = (const float*)d_in[10];
  const float* eWihB = (const float*)d_in[11];
  const float* eWhhB = (const float*)d_in[12];
  const float* eBihB = (const float*)d_in[13];
  const float* eBhhB = (const float*)d_in[14];
  const float* w_mu  = (const float*)d_in[15];
  const float* b_mu  = (const float*)d_in[16];
  const float* w_var = (const float*)d_in[17];
  const float* b_var = (const float*)d_in[18];
  const float* w_dr  = (const float*)d_in[19];
  const float* b_dr  = (const float*)d_in[20];
  const float* w_dn  = (const float*)d_in[21];
  const float* b_dn  = (const float*)d_in[22];
  const float* w_init= (const float*)d_in[23];
  const float* b_init= (const float*)d_in[24];
  const float* g1Wih = (const float*)d_in[25];
  const float* g1Whh = (const float*)d_in[26];
  const float* g1Bih = (const float*)d_in[27];
  const float* g1Bhh = (const float*)d_in[28];
  const float* g2Wih = (const float*)d_in[29];
  const float* g2Whh = (const float*)d_in[30];
  const float* g2Bih = (const float*)d_in[31];
  const float* g2Bhh = (const float*)d_in[32];
  const float* w_out = (const float*)d_in[33];
  const float* b_out = (const float*)d_in[34];

  // Output tuple, concatenated flat in return order.
  float* o = (float*)d_out;
  float* o_main = o;
  float* o_r    = o_main + (size_t)kB * kT * kRoll;
  float* o_n    = o_r + kB;
  float* o_mu   = o_n + kB;
  float* o_std  = o_mu + (size_t)kB * kZd;
  float* o_z    = o_std + (size_t)kB * kZd;

  // Workspace carve (~16 MB), 256B aligned slots.
  uint8_t* wp = (uint8_t*)d_ws;
  auto carve = [&](size_t bytes) -> void* {
    void* p = (void*)wp;
    wp += (bytes + 255) & ~(size_t)255;
    return p;
  };
  _Float16* WihF16 = (_Float16*)carve((size_t)kG * kXS * 2);
  _Float16* WihB16 = (_Float16*)carve((size_t)kG * kXS * 2);
  _Float16* WhhF16 = (_Float16*)carve((size_t)kG * kHid * 2);
  _Float16* WhhB16 = (_Float16*)carve((size_t)kG * kHid * 2);
  _Float16* G1hh16 = (_Float16*)carve((size_t)kG * kHid * 2);
  _Float16* G2ih16 = (_Float16*)carve((size_t)kG * kHid * 2);
  _Float16* G2hh16 = (_Float16*)carve((size_t)kG * kHid * 2);
  _Float16* G1T16  = (_Float16*)carve((size_t)kRoll * kG * 2);
  _Float16* Wout16 = (_Float16*)carve((size_t)kXS * kHid * 2);
  float* hf   = (float*)carve((size_t)kB * kHid * 4);
  float* hb   = (float*)carve((size_t)kB * kHid * 4);
  float* Zc   = (float*)carve((size_t)kB * kG * 4);
  float* hg10 = (float*)carve((size_t)kB * kHid * 4);

  const int th = 256;
  auto blocks = [](int total) { return (total + 255) / 256; };

  // 1) weight staging
  cvt_pad_kernel<<<blocks(kG * kXS), th, 0, stream>>>(WihF16, eWihF, kG, kRoll, kXS, kG);
  cvt_pad_kernel<<<blocks(kG * kXS), th, 0, stream>>>(WihB16, eWihB, kG, kRoll, kXS, kG);
  cvt_pad_kernel<<<blocks(kG * kHid), th, 0, stream>>>(WhhF16, eWhhF, kG, kHid, kHid, kG);
  cvt_pad_kernel<<<blocks(kG * kHid), th, 0, stream>>>(WhhB16, eWhhB, kG, kHid, kHid, kG);
  cvt_pad_kernel<<<blocks(kG * kHid), th, 0, stream>>>(G1hh16, g1Whh, kG, kHid, kHid, kG);
  cvt_pad_kernel<<<blocks(kG * kHid), th, 0, stream>>>(G2ih16, g2Wih, kG, kHid, kHid, kG);
  cvt_pad_kernel<<<blocks(kG * kHid), th, 0, stream>>>(G2hh16, g2Whh, kG, kHid, kHid, kG);
  cvt_pad_kernel<<<blocks(kXS * kHid), th, 0, stream>>>(Wout16, w_out, kRoll, kHid, kHid, kXS);
  g1_transpose_kernel<<<blocks(kRoll * kG), th, 0, stream>>>(G1T16, g1Wih);

  // 2) bidirectional encoder (independent; only final hidden states needed)
  enc_kernel<<<kB / 16, 512, 0, stream>>>(x, WihF16, WhhF16, eBihF, eBhhF, hf, 0);
  enc_kernel<<<kB / 16, 512, 0, stream>>>(x, WihB16, WhhB16, eBihB, eBhhB, hb, 1);

  // 3) latent heads + decoder constants
  latent_kernel<<<kB, 256, 0, stream>>>(hf, hb, w_mu, b_mu, w_var, b_var,
                                        w_dr, b_dr, w_dn, b_dn, w_init, b_init,
                                        g1Wih, g1Bih, eps, r_density, n_density,
                                        o_mu, o_std, o_z, o_r, o_n, Zc, hg10);

  // 4) autoregressive decoder
  dec_kernel<<<kB / 16, 512, 0, stream>>>(G1hh16, G2ih16, G2hh16, G1T16, Wout16,
                                          Zc, hg10, g1Bhh, g2Bih, g2Bhh, b_out,
                                          o_main);
}
